// Transformer_18906446037187
// MI455X (gfx1250) — matrix operand order repaired and checked by reading, not compiled, not run on hardware
//
#include <hip/hip_runtime.h>
#include <hip/hip_bf16.h>
#include <math.h>

// ---------------------------------------------------------------------------
// CDNA5 (gfx1250) Transformer forward.  All GEMMs run on the matrix cores via
// v_wmma_f32_16x16x32_bf16 (wave32).  All GEMM operands are pre-converted to
// bf16 exactly once (weights by a cvt pass, activations by their producer
// kernels), so the GEMM hot loop is pure global_load_b128 -> WMMA with no
// conversion VALU and no predication: K is always a multiple of 32 (P rows
// padded to 416 with zeros) and M/N edges are handled by index clamping
// (garbage lands only in rows/cols the epilogue discards).
//
// TB (weight) GEMMs: 128x128 block tile, 8 waves as 4x2, wave tile 32x64
//   -> 8 WMMAs per K-step from 12 b128 loads.
// NN (P@V) GEMMs: 64x128 block tile, 8 waves as 2x4, wave tile 32x32.
// ---------------------------------------------------------------------------

typedef __attribute__((ext_vector_type(16))) __bf16 v16bf;
typedef __attribute__((ext_vector_type(8)))  float  v8f;

__device__ __forceinline__ unsigned short f32_to_bf16_rne(float f) {
  union { float f; unsigned int u; } c; c.f = f;
  unsigned int u = c.u;
  u += 0x7FFFu + ((u >> 16) & 1u);     // round to nearest even
  return (unsigned short)(u >> 16);
}

union FragBf { v16bf v; uint4 q[2]; unsigned short u[16]; };

// A fragment: per lane 16 contiguous K values
// (pointer pre-offset by 16*laneHalf).  Two b128 loads.
__device__ __forceinline__ void load_a_frag(FragBf& f, const unsigned short* p) {
  f.q[0] = *(const uint4*)(p);
  f.q[1] = *(const uint4*)(p + 8);
}

// B fragment (32x16, same k order as A): per lane 16 contiguous K values
// (pointer pre-offset by 16*laneHalf).  Two b128 loads.
__device__ __forceinline__ void load_b_frag(FragBf& f, const unsigned short* p) {
  f.q[0] = *(const uint4*)(p);
  f.q[1] = *(const uint4*)(p + 8);
}

struct GemmP {
  const unsigned short* A;   // bf16 [M,K], row stride lda (elements)
  const unsigned short* B;   // bf16; TB=1: [N,K] stride ldb ; TB=0: [K,N] stride ldb
  float*          C32;       // optional fp32 out [M,N] stride ldc
  unsigned short* C16;       // optional bf16 out [M,N] stride ldc
  const float*    bias;      // [N] or null
  int  M, N, K, KB, lda, ldb, ldc;           // K multiple of 32; KB = valid B rows (NN path)
  long sAo, sAi, sBo, sBi, sCo, sCi, sBias;  // batch strides (z = zo*binner + zi)
  int  binner, act;                          // act: 0 none, 1 relu, 2 sigmoid
  float alpha;
};

template <bool TB>
__global__ __launch_bounds__(256) void gemm_bf16_wmma(GemmP p) {
  constexpr int NI = TB ? 4 : 2;            // B fragments per wave
  constexpr int BM = TB ? 128 : 64;         // block tile M
  constexpr int BN = 128;                   // block tile N

  const int z  = blockIdx.z;
  const int zo = z / p.binner, zi = z % p.binner;
  const unsigned short* A = p.A + zo * p.sAo + zi * p.sAi;
  const unsigned short* B = p.B + zo * p.sBo + zi * p.sBi;
  const float* bias = p.bias ? (p.bias + (long)z * p.sBias) : (const float*)0;

  const int lane = threadIdx.x & 31;
  const int wid  = threadIdx.x >> 5;
  const int wm   = TB ? (wid & 3) : (wid & 1);
  const int wn   = TB ? (wid >> 2) : (wid >> 1);
  const int m0   = blockIdx.y * BM + wm * 32;
  const int n0   = blockIdx.x * BN + wn * (16 * NI);
  const int l15  = lane & 15;
  const int lh   = lane >> 4;               // lane half

  // Clamped per-lane row/col bases (edge garbage discarded by epilogue).
  const int rA0 = min(m0 + l15,      p.M - 1);
  const int rA1 = min(m0 + 16 + l15, p.M - 1);
  const unsigned short* aP0 = A + (long)rA0 * p.lda + 16 * lh;
  const unsigned short* aP1 = A + (long)rA1 * p.lda + 16 * lh;

  int cB[NI];
  #pragma unroll
  for (int ni = 0; ni < NI; ++ni) cB[ni] = min(n0 + 16 * ni + l15, p.N - 1);

  v8f acc[2][NI] = {};

  if (TB) {
    const unsigned short* bP[NI];
    #pragma unroll
    for (int ni = 0; ni < NI; ++ni) bP[ni] = B + (long)cB[ni] * p.ldb + 16 * lh;
    for (int k0 = 0; k0 < p.K; k0 += 32) {
      FragBf a[2], b[NI];
      load_a_frag(a[0], aP0 + k0);
      load_a_frag(a[1], aP1 + k0);
      #pragma unroll
      for (int ni = 0; ni < NI; ++ni) load_b_frag(b[ni], bP[ni] + k0);
      #pragma unroll
      for (int mi = 0; mi < 2; ++mi)
        #pragma unroll
        for (int ni = 0; ni < NI; ++ni)
          acc[mi][ni] = __builtin_amdgcn_wmma_f32_16x16x32_bf16(
              false, a[mi].v, false, b[ni].v, (short)0, acc[mi][ni], false, false);
    }
  } else {
    for (int k0 = 0; k0 < p.K; k0 += 32) {
      FragBf a[2], b[NI];
      load_a_frag(a[0], aP0 + k0);
      load_a_frag(a[1], aP1 + k0);
      #pragma unroll
      for (int j = 0; j < 16; ++j) {
        int kk = k0 + 16 * lh + j;
        kk = kk < p.KB ? kk : p.KB - 1;     // clamp: P there is 0 anyway
        const unsigned short* bp = B + (long)kk * p.ldb;
        #pragma unroll
        for (int ni = 0; ni < NI; ++ni) b[ni].u[j] = bp[cB[ni]];
      }
      #pragma unroll
      for (int mi = 0; mi < 2; ++mi)
        #pragma unroll
        for (int ni = 0; ni < NI; ++ni)
          acc[mi][ni] = __builtin_amdgcn_wmma_f32_16x16x32_bf16(
              false, a[mi].v, false, b[ni].v, (short)0, acc[mi][ni], false, false);
    }
  }

  // Epilogue (C/D layout: VGPR r -> M = r + 8*laneHalf, N = lane&15)
  float*          C32 = p.C32 ? p.C32 + zo * p.sCo + zi * p.sCi : (float*)0;
  unsigned short* C16 = p.C16 ? p.C16 + zo * p.sCo + zi * p.sCi : (unsigned short*)0;
  #pragma unroll
  for (int mi = 0; mi < 2; ++mi) {
    #pragma unroll
    for (int ni = 0; ni < NI; ++ni) {
      const int col = n0 + 16 * ni + l15;
      if (col >= p.N) continue;
      const float bv = bias ? bias[col] : 0.0f;
      #pragma unroll
      for (int r = 0; r < 8; ++r) {
        const int row = m0 + 16 * mi + 8 * lh + r;
        if (row >= p.M) continue;
        float v = acc[mi][ni][r] * p.alpha + bv;
        if (p.act == 1)      v = fmaxf(v, 0.0f);
        else if (p.act == 2) v = 1.0f / (1.0f + __expf(-v));
        const long idx = (long)row * p.ldc + col;
        if (C32) C32[idx] = v;
        if (C16) C16[idx] = f32_to_bf16_rne(v);
      }
    }
  }
}

// ---------------------------------------------------------------------------
// LayerNorm: o = LN(in1 + in2) * s + b.  One wave32 per row (width 512).
// Writes fp32 (residual stream) and/or bf16 (next GEMM A operand).
// ---------------------------------------------------------------------------
__global__ __launch_bounds__(256) void ln_kernel(const float* in1, const float* in2,
                                                 float* out32, unsigned short* out16,
                                                 const float* s, const float* b,
                                                 int rows, int width) {
  const int lane = threadIdx.x & 31;
  const int wid  = threadIdx.x >> 5;
  const long row = (long)blockIdx.x * 8 + wid;
  if (row >= rows) return;
  const float* p1 = in1 + row * width;
  const float* p2 = in2 ? in2 + row * width : (const float*)0;
  float sum = 0.f, sq = 0.f;
  for (int i = lane; i < width; i += 32) {
    float v = p1[i] + (p2 ? p2[i] : 0.f);
    sum += v; sq += v * v;
  }
  #pragma unroll
  for (int o = 16; o > 0; o >>= 1) {
    sum += __shfl_xor(sum, o, 32);
    sq  += __shfl_xor(sq,  o, 32);
  }
  const float mean = sum / width;
  const float var  = sq / width - mean * mean;
  const float r    = rsqrtf(var + 1e-5f);
  for (int i = lane; i < width; i += 32) {
    float v = p1[i] + (p2 ? p2[i] : 0.f);
    float o = (v - mean) * r * s[i] + b[i];
    if (out32) out32[row * width + i] = o;
    if (out16) out16[row * width + i] = f32_to_bf16_rne(o);
  }
}

// ---------------------------------------------------------------------------
// Softmax: reads fp32 scores (row stride ldin), writes bf16 probabilities
// (row stride ldout, zero-padded in [lim, ldout)).  One wave per row.
// ---------------------------------------------------------------------------
__global__ __launch_bounds__(256) void softmax_kernel(const float* Sin, unsigned short* Sout,
                                                      long totalRows, int qlen, int cols,
                                                      int ldin, int ldout, int causal) {
  const int lane = threadIdx.x & 31;
  const int wid  = threadIdx.x >> 5;
  const long row = (long)blockIdx.x * 8 + wid;
  if (row >= totalRows) return;
  const int q   = (int)(row % qlen);
  const int lim = causal ? (q + 1) : cols;
  const float* p = Sin + row * (long)ldin;
  unsigned short* o = Sout + row * (long)ldout;
  float mx = -INFINITY;
  for (int i = lane; i < lim; i += 32) mx = fmaxf(mx, p[i]);
  #pragma unroll
  for (int off = 16; off > 0; off >>= 1) mx = fmaxf(mx, __shfl_xor(mx, off, 32));
  float sum = 0.f;
  for (int i = lane; i < lim; i += 32) sum += __expf(p[i] - mx);
  #pragma unroll
  for (int off = 16; off > 0; off >>= 1) sum += __shfl_xor(sum, off, 32);
  const float inv = 1.0f / sum;
  for (int i = lane; i < lim; i += 32) o[i] = f32_to_bf16_rne(__expf(p[i] - mx) * inv);
  for (int i = lim + lane; i < ldout; i += 32) o[i] = 0;
}

// ---------------------------------------------------------------------------
// Elementwise helpers
// ---------------------------------------------------------------------------
__global__ void cvt_bf16_kernel(const float* in, unsigned short* out, long n) {
  long i = (long)blockIdx.x * blockDim.x + threadIdx.x;
  if (i < n) out[i] = f32_to_bf16_rne(in[i]);
}

__global__ void transpose_src_kernel(const float* src, unsigned short* dst,
                                     int N_, int E_, int S_) {
  long i = (long)blockIdx.x * blockDim.x + threadIdx.x;
  const long total = (long)N_ * E_ * S_;
  if (i >= total) return;
  const int s = (int)(i % S_);
  const long t = i / S_;
  const int e = (int)(t % E_);
  const int n = (int)(t / E_);
  dst[((long)s * N_ + n) * E_ + e] = f32_to_bf16_rne(src[i]); // dst[s][n][e]=src[n][e][s]
}

__global__ void add_pos_kernel(float* x, unsigned short* xb, const float* pos,
                               long total, long SE) {
  long i = (long)blockIdx.x * blockDim.x + threadIdx.x;
  if (i >= total) return;
  float v = x[i] + pos[i % SE];
  x[i] = v;
  xb[i] = f32_to_bf16_rne(v);
}

__global__ void dec_embed_kernel(const int* trg, const float* trg_word,
                                 const float* pos_trg, float* y, unsigned short* yb,
                                 int N_, int T_, int E_) {
  long i = (long)blockIdx.x * blockDim.x + threadIdx.x;
  const long total = (long)N_ * T_ * E_;
  if (i >= total) return;
  const int e = (int)(i % E_);
  const long t2 = i / E_;
  const int t = (int)(t2 % T_);
  const int n = (int)(t2 / T_);
  const int tok = trg[(long)n * T_ + t];
  float v = trg_word[(long)tok * E_ + e] + pos_trg[(long)t * E_ + e];
  y[i] = v;
  yb[i] = f32_to_bf16_rne(v);
}

// ---------------------------------------------------------------------------
// Host orchestration
// ---------------------------------------------------------------------------
static inline void gemm(hipStream_t st,
                        const unsigned short* A, int lda, long sAo, long sAi,
                        const unsigned short* B, int ldb, long sBo, long sBi, int transB,
                        float* C32, unsigned short* C16, int ldc, long sCo, long sCi,
                        const float* bias, long sBias,
                        int M, int N, int K, int KB, int batch, int binner,
                        float alpha, int act) {
  GemmP p;
  p.A = A; p.B = B; p.C32 = C32; p.C16 = C16; p.bias = bias;
  p.M = M; p.N = N; p.K = K; p.KB = KB; p.lda = lda; p.ldb = ldb; p.ldc = ldc;
  p.sAo = sAo; p.sAi = sAi; p.sBo = sBo; p.sBi = sBi; p.sCo = sCo; p.sCi = sCi;
  p.sBias = sBias; p.binner = binner; p.act = act; p.alpha = alpha;
  const int BMv = transB ? 128 : 64;
  dim3 g((N + 127) / 128, (M + BMv - 1) / BMv, batch);
  if (transB) gemm_bf16_wmma<true ><<<g, dim3(256), 0, st>>>(p);
  else        gemm_bf16_wmma<false><<<g, dim3(256), 0, st>>>(p);
}

extern "C" void kernel_launch(void* const* d_in, const int* in_sizes, int n_in,
                              void* d_out, int out_size, void* d_ws, size_t ws_size,
                              hipStream_t stream) {
  (void)in_sizes; (void)n_in; (void)out_size; (void)ws_size;
  const int Nb = 16, E = 512, S = 400, T = 400, F = 2048, NH = 8, Le = 3, Ld = 3;
  const int SLD = 416;                     // padded bf16 P row stride (13*32)
  const long NSE = (long)Nb * S * E;       // 3,276,800 elems
  const long SE  = (long)S * E;

  int ii = 0;
  const float* src       = (const float*)d_in[ii++];
  const int*   trg       = (const int*)  d_in[ii++];
  const float* W_emb     = (const float*)d_in[ii++];
  const float* b_emb     = (const float*)d_in[ii++];
  const float* pos_src   = (const float*)d_in[ii++];
  const float* pos_trg   = (const float*)d_in[ii++];
  const float* trg_word  = (const float*)d_in[ii++];
  const float* enc_qkv_w = (const float*)d_in[ii++];
  const float* enc_qkv_b = (const float*)d_in[ii++];
  const float* enc_out_w = (const float*)d_in[ii++];
  const float* enc_out_b = (const float*)d_in[ii++];
  const float* enc_ln1_s = (const float*)d_in[ii++];
  const float* enc_ln1_b = (const float*)d_in[ii++];
  const float* enc_ln2_s = (const float*)d_in[ii++];
  const float* enc_ln2_b = (const float*)d_in[ii++];
  const float* enc_ff1_w = (const float*)d_in[ii++];
  const float* enc_ff1_b = (const float*)d_in[ii++];
  const float* enc_ff2_w = (const float*)d_in[ii++];
  const float* enc_ff2_b = (const float*)d_in[ii++];
  const float* enc_norm_s= (const float*)d_in[ii++];
  const float* enc_norm_b= (const float*)d_in[ii++];
  const float* dec_qkv_w = (const float*)d_in[ii++];
  const float* dec_qkv_b = (const float*)d_in[ii++];
  const float* dec_sa_out_w = (const float*)d_in[ii++];
  const float* dec_sa_out_b = (const float*)d_in[ii++];
  const float* dec_caq_w = (const float*)d_in[ii++];
  const float* dec_caq_b = (const float*)d_in[ii++];
  const float* dec_cakv_w= (const float*)d_in[ii++];
  const float* dec_cakv_b= (const float*)d_in[ii++];
  const float* dec_ca_out_w = (const float*)d_in[ii++];
  const float* dec_ca_out_b = (const float*)d_in[ii++];
  const float* dec_ln1_s = (const float*)d_in[ii++];
  const float* dec_ln1_b = (const float*)d_in[ii++];
  const float* dec_ln2_s = (const float*)d_in[ii++];
  const float* dec_ln2_b = (const float*)d_in[ii++];
  const float* dec_ln3_s = (const float*)d_in[ii++];
  const float* dec_ln3_b = (const float*)d_in[ii++];
  const float* dec_ff1_w = (const float*)d_in[ii++];
  const float* dec_ff1_b = (const float*)d_in[ii++];
  const float* dec_ff2_w = (const float*)d_in[ii++];
  const float* dec_ff2_b = (const float*)d_in[ii++];
  const float* dec_norm_s= (const float*)d_in[ii++];
  const float* dec_norm_b= (const float*)d_in[ii++];
  const float* fc_w      = (const float*)d_in[ii++];
  const float* fc_b      = (const float*)d_in[ii++];
  float* out = (float*)d_out;

  // --------- Workspace carve: fp32 region then bf16 region (~525 MB) -------
  float* wp = (float*)d_ws;
  float* x      = wp; wp += NSE;                    // [N,S,E] residual stream
  float* y      = wp; wp += NSE;                    // [N,T,E] residual stream
  float* proj   = wp; wp += NSE;                    // [N,S,E] sublayer output
  float* scores = wp; wp += (long)Nb * NH * S * S;  // [N*NH,S,S] fp32, ld=400

  unsigned short* sp = (unsigned short*)wp;
  #define TAKE(name, n) unsigned short* name = sp; sp += (n)
  TAKE(srcTb,  NSE);                      // [S,N,E]
  TAKE(xb,     NSE);
  TAKE(yb,     NSE);
  TAKE(memb,   NSE);
  TAKE(attnb,  NSE);
  TAKE(ynb,    NSE);
  TAKE(qkvb,   (long)Nb * S * 3 * E);     // also reused as q
  TAKE(kvb,    (long)Nb * S * 2 * E);
  TAKE(ffb,    (long)Nb * S * F);
  TAKE(scoresb,(long)Nb * NH * S * SLD);  // bf16 P, ld=416, zero-padded
  TAKE(wembB,  (long)S * E * E);
  TAKE(eqkvB,  (long)Le * 3 * E * E);
  TAKE(eoutB,  (long)Le * E * E);
  TAKE(eff1B,  (long)Le * F * E);
  TAKE(eff2B,  (long)Le * E * F);
  TAKE(dqkvB,  (long)Ld * 3 * E * E);
  TAKE(dsaoB,  (long)Ld * E * E);
  TAKE(dcaqB,  (long)Ld * E * E);
  TAKE(dcakvB, (long)Ld * 2 * E * E);
  TAKE(dcaoB,  (long)Ld * E * E);
  TAKE(dff1B,  (long)Ld * F * E);
  TAKE(dff2B,  (long)Ld * E * F);
  TAKE(fcB,    (long)512 * E);
  #undef TAKE

  const int rowsNS = Nb * S;                        // 6400
  const float iscale = 0.125f;                      // 1/sqrt(64)

  // ---- one-time (per call) weight conversion to bf16 ----
  auto cvt = [&](const float* in, unsigned short* o, long n) {
    cvt_bf16_kernel<<<(unsigned)((n + 255) / 256), 256, 0, stream>>>(in, o, n);
  };
  cvt(W_emb,     wembB,  (long)S * E * E);
  cvt(enc_qkv_w, eqkvB,  (long)Le * 3 * E * E);
  cvt(enc_out_w, eoutB,  (long)Le * E * E);
  cvt(enc_ff1_w, eff1B,  (long)Le * F * E);
  cvt(enc_ff2_w, eff2B,  (long)Le * E * F);
  cvt(dec_qkv_w, dqkvB,  (long)Ld * 3 * E * E);
  cvt(dec_sa_out_w, dsaoB, (long)Ld * E * E);
  cvt(dec_caq_w, dcaqB,  (long)Ld * E * E);
  cvt(dec_cakv_w, dcakvB,(long)Ld * 2 * E * E);
  cvt(dec_ca_out_w, dcaoB,(long)Ld * E * E);
  cvt(dec_ff1_w, dff1B,  (long)Ld * F * E);
  cvt(dec_ff2_w, dff2B,  (long)Ld * E * F);
  cvt(fc_w,      fcB,    (long)512 * E);

  // ---- Embedding: srcT[s] @ W_emb[s]^T, sigmoid, + pos ----
  {
    long tot = (long)Nb * E * S;
    transpose_src_kernel<<<(unsigned)((tot + 255) / 256), 256, 0, stream>>>(src, srcTb, Nb, E, S);
  }
  gemm(stream, srcTb, E, (long)Nb * E, 0,
       wembB, E, (long)E * E, 0, 1,
       x, (unsigned short*)0, (int)SE, (long)E, 0,  // C[n][s=z][f]: ldc=S*E, batch off = z*E
       b_emb, (long)E,
       Nb, E, E, E, S, 1, 1.0f, 2);
  add_pos_kernel<<<(unsigned)((NSE + 255) / 256), 256, 0, stream>>>(x, xb, pos_src, NSE, SE);

  // ---- Decoder input embedding ----
  dec_embed_kernel<<<(unsigned)((NSE + 255) / 256), 256, 0, stream>>>(trg, trg_word, pos_trg, y, yb, Nb, T, E);

  const long sQKV  = (long)S * 3 * E;      // per-b stride in qkvb
  const long sSCf  = (long)S * S;          // per-h stride in fp32 scores (ld 400)
  const long sSCfo = (long)NH * S * S;
  const long sSCb  = (long)S * SLD;        // per-h stride in bf16 P (ld 416)
  const long sSCbo = (long)NH * S * SLD;
  const long ROWSM = (long)S * E;          // per-b stride in [N,S,E]
  const long sKV   = (long)S * 2 * E;

  // ================= Encoder =================
  for (int i = 0; i < Le; ++i) {
    gemm(stream, xb, E, 0, 0, eqkvB + (long)i * 3 * E * E, E, 0, 0, 1,
         (float*)0, qkvb, 3 * E, 0, 0, enc_qkv_b + (long)i * 3 * E, 0,
         rowsNS, 3 * E, E, E, 1, 1, 1.0f, 0);
    gemm(stream, qkvb, 3 * E, sQKV, 64,
         qkvb + E, 3 * E, sQKV, 64, 1,
         scores, (unsigned short*)0, S, sSCfo, sSCf, (const float*)0, 0,
         S, S, 64, 64, Nb * NH, NH, iscale, 0);
    softmax_kernel<<<(unsigned)(((long)Nb * NH * S + 7) / 8), 256, 0, stream>>>(
        scores, scoresb, (long)Nb * NH * S, S, S, S, SLD, 0);
    gemm(stream, scoresb, SLD, sSCbo, sSCb,
         qkvb + 2 * E, 3 * E, sQKV, 64, 0,
         (float*)0, attnb, E, ROWSM, 64, (const float*)0, 0,
         S, 64, SLD, S, Nb * NH, NH, 1.0f, 0);
    gemm(stream, attnb, E, 0, 0, eoutB + (long)i * E * E, E, 0, 0, 1,
         proj, (unsigned short*)0, E, 0, 0, enc_out_b + (long)i * E, 0,
         rowsNS, E, E, E, 1, 1, 1.0f, 0);
    ln_kernel<<<(rowsNS + 7) / 8, 256, 0, stream>>>(x, proj, x, xb,
        enc_ln1_s + (long)i * E, enc_ln1_b + (long)i * E, rowsNS, E);
    gemm(stream, xb, E, 0, 0, eff1B + (long)i * F * E, E, 0, 0, 1,
         (float*)0, ffb, F, 0, 0, enc_ff1_b + (long)i * F, 0,
         rowsNS, F, E, E, 1, 1, 1.0f, 1);
    gemm(stream, ffb, F, 0, 0, eff2B + (long)i * E * F, F, 0, 0, 1,
         proj, (unsigned short*)0, E, 0, 0, enc_ff2_b + (long)i * E, 0,
         rowsNS, E, F, F, 1, 1, 1.0f, 0);
    ln_kernel<<<(rowsNS + 7) / 8, 256, 0, stream>>>(x, proj, x, xb,
        enc_ln2_s + (long)i * E, enc_ln2_b + (long)i * E, rowsNS, E);
  }
  ln_kernel<<<(rowsNS + 7) / 8, 256, 0, stream>>>(x, (const float*)0, (float*)0, memb,
      enc_norm_s, enc_norm_b, rowsNS, E);

  // ================= Decoder =================
  for (int i = 0; i < Ld; ++i) {
    // --- causal self-attention ---
    gemm(stream, yb, E, 0, 0, dqkvB + (long)i * 3 * E * E, E, 0, 0, 1,
         (float*)0, qkvb, 3 * E, 0, 0, dec_qkv_b + (long)i * 3 * E, 0,
         rowsNS, 3 * E, E, E, 1, 1, 1.0f, 0);
    gemm(stream, qkvb, 3 * E, sQKV, 64,
         qkvb + E, 3 * E, sQKV, 64, 1,
         scores, (unsigned short*)0, T, sSCfo, sSCf, (const float*)0, 0,
         T, T, 64, 64, Nb * NH, NH, iscale, 0);
    softmax_kernel<<<(unsigned)(((long)Nb * NH * T + 7) / 8), 256, 0, stream>>>(
        scores, scoresb, (long)Nb * NH * T, T, T, T, SLD, 1);
    gemm(stream, scoresb, SLD, sSCbo, sSCb,
         qkvb + 2 * E, 3 * E, sQKV, 64, 0,
         (float*)0, attnb, E, ROWSM, 64, (const float*)0, 0,
         T, 64, SLD, T, Nb * NH, NH, 1.0f, 0);
    gemm(stream, attnb, E, 0, 0, dsaoB + (long)i * E * E, E, 0, 0, 1,
         proj, (unsigned short*)0, E, 0, 0, dec_sa_out_b + (long)i * E, 0,
         rowsNS, E, E, E, 1, 1, 1.0f, 0);
    ln_kernel<<<(rowsNS + 7) / 8, 256, 0, stream>>>(y, proj, y, yb,
        dec_ln1_s + (long)i * E, dec_ln1_b + (long)i * E, rowsNS, E);
    // --- cross-attention ---
    gemm(stream, yb, E, 0, 0, dcaqB + (long)i * E * E, E, 0, 0, 1,
         (float*)0, qkvb, E, 0, 0, dec_caq_b + (long)i * E, 0,   // q -> reuse qkvb [N*T,E]
         rowsNS, E, E, E, 1, 1, 1.0f, 0);
    gemm(stream, memb, E, 0, 0, dcakvB + (long)i * 2 * E * E, E, 0, 0, 1,
         (float*)0, kvb, 2 * E, 0, 0, dec_cakv_b + (long)i * 2 * E, 0,
         rowsNS, 2 * E, E, E, 1, 1, 1.0f, 0);
    gemm(stream, qkvb, E, ROWSM, 64,
         kvb, 2 * E, sKV, 64, 1,
         scores, (unsigned short*)0, S, sSCfo, sSCf, (const float*)0, 0,
         T, S, 64, 64, Nb * NH, NH, iscale, 0);
    softmax_kernel<<<(unsigned)(((long)Nb * NH * T + 7) / 8), 256, 0, stream>>>(
        scores, scoresb, (long)Nb * NH * T, T, S, S, SLD, 0);
    gemm(stream, scoresb, SLD, sSCbo, sSCb,
         kvb + E, 2 * E, sKV, 64, 0,
         (float*)0, attnb, E, ROWSM, 64, (const float*)0, 0,
         T, 64, SLD, S, Nb * NH, NH, 1.0f, 0);
    gemm(stream, attnb, E, 0, 0, dcaoB + (long)i * E * E, E, 0, 0, 1,
         proj, (unsigned short*)0, E, 0, 0, dec_ca_out_b + (long)i * E, 0,
         rowsNS, E, E, E, 1, 1, 1.0f, 0);
    ln_kernel<<<(rowsNS + 7) / 8, 256, 0, stream>>>(y, proj, y, yb,
        dec_ln2_s + (long)i * E, dec_ln2_b + (long)i * E, rowsNS, E);
    // --- FFN ---
    gemm(stream, yb, E, 0, 0, dff1B + (long)i * F * E, E, 0, 0, 1,
         (float*)0, ffb, F, 0, 0, dec_ff1_b + (long)i * F, 0,
         rowsNS, F, E, E, 1, 1, 1.0f, 1);
    gemm(stream, ffb, F, 0, 0, dff2B + (long)i * E * F, F, 0, 0, 1,
         proj, (unsigned short*)0, E, 0, 0, dec_ff2_b + (long)i * E, 0,
         rowsNS, E, F, F, 1, 1, 1.0f, 0);
    ln_kernel<<<(rowsNS + 7) / 8, 256, 0, stream>>>(y, proj, y, yb,
        dec_ln3_s + (long)i * E, dec_ln3_b + (long)i * E, rowsNS, E);
  }

  // final norm + logits
  ln_kernel<<<(rowsNS + 7) / 8, 256, 0, stream>>>(y, (const float*)0, (float*)0, ynb,
      dec_norm_s, dec_norm_b, rowsNS, E);
  gemm(stream, ynb, E, 0, 0, fcB, E, 0, 0, 1,
       out, (unsigned short*)0, 512, 0, 0, fc_b, 0,
       rowsNS, 512, E, E, 1, 1, 1.0f, 0);
}
